// CAConv_63728724738332
// MI455X (gfx1250) — matrix operand order repaired and checked by reading, not compiled, not run on hardware
//
#include <hip/hip_runtime.h>

typedef __attribute__((ext_vector_type(16))) __bf16        v16bf;
typedef __attribute__((ext_vector_type(8)))  float         v8f;
typedef __attribute__((ext_vector_type(4)))  unsigned int  v4u;

#define B_    32
#define CIN_  64
#define COUT_ 64
#define H_    128
#define W_    128
#define KTOT  576            // 9 taps * 64 cin, K ordered as tap*64 + ci
#define CLEN  8
#define HID   16

__device__ __forceinline__ unsigned short f2bf(float f) {
    unsigned int u = __float_as_uint(f);
    u += 0x7fffu + ((u >> 16) & 1u);     // round-to-nearest-even
    return (unsigned short)(u >> 16);
}

// ---------------------------------------------------------------------------
// Stage 1: per-sample weights  W_b = conv_w + relu(c W1 + b1) W2 + b2,
// written bf16, reordered so GEMM-K = tap*64 + ci (tap = kh*3+kw).
// One block per (b, cout); 192 threads * 3 iters = 576 K entries.
// ---------------------------------------------------------------------------
__global__ void caconv_wgen(const float* __restrict__ c,
                            const float* __restrict__ conv_w,
                            const float* __restrict__ w1,
                            const float* __restrict__ b1,
                            const float* __restrict__ w2,
                            const float* __restrict__ b2,
                            unsigned short* __restrict__ wr) {
    const int b  = blockIdx.x >> 6;
    const int co = blockIdx.x & 63;

    float h[HID];
#pragma unroll
    for (int m = 0; m < HID; ++m) {
        float a = b1[m];
#pragma unroll
        for (int k = 0; k < CLEN; ++k) a += c[b * CLEN + k] * w1[k * HID + m];
        h[m] = a > 0.0f ? a : 0.0f;
    }

    for (int i = threadIdx.x; i < KTOT; i += blockDim.x) {
        const int tap = i >> 6;          // kh*3+kw
        const int ci  = i & 63;
        const int j   = (co * 64 + ci) * 9 + tap;   // flat [cout][ci][kh][kw]
        float v = conv_w[j] + b2[j];
#pragma unroll
        for (int m = 0; m < HID; ++m) v += h[m] * w2[m * (COUT_ * CIN_ * 9) + j];
        wr[(b * COUT_ + co) * KTOT + i] = f2bf(v);
    }
}

// ---------------------------------------------------------------------------
// Stage 2: implicit-GEMM conv with v_wmma_f32_16x16x32_bf16.
// Block = (sample b, output row y). 8 waves: wave w -> cout tile (w&3),
// pixel half (w>>2); each wave holds 4 16x16 f32 accumulators (64 pixels).
// LDS tile: 3 input rows x 130 padded cols x 64 ci, bf16, layout [r][col][ci]
// so each B-fragment lane reads 16 contiguous ci values (32B -> ds_load_b128).
// A (global) and B (LDS) fragments are double-buffered across K-steps so the
// 4 WMMAs of step s overlap the loads of step s+1 instead of waiting on
// just-issued loads.
// ---------------------------------------------------------------------------
__global__ void __launch_bounds__(256)
caconv_main(const float* __restrict__ x,
            const unsigned short* __restrict__ wr,
            const float* __restrict__ conv_b,
            float* __restrict__ out) {
    __shared__ __align__(32) unsigned short xs[3 * 130 * 64];   // 49,920 B

    const int y = blockIdx.x;
    const int b = blockIdx.y;

    // ---- stage input rows y-1..y+1 into LDS as bf16 (coalesced along W) ----
    for (int i = threadIdx.x; i < 3 * 130 * 64; i += 256) {
        const int col = i % 130;                 // padded column (input col-1)
        const int ci  = (i / 130) & 63;
        const int r   = i / (130 * 64);          // 0..2 -> input row y-1+r
        const int yy  = y - 1 + r;
        const int xx  = col - 1;
        float v = 0.0f;
        if (yy >= 0 && yy < H_ && xx >= 0 && xx < W_)
            v = x[((size_t)(b * CIN_ + ci) * H_ + yy) * W_ + xx];
        xs[(r * 130 + col) * 64 + ci] = f2bf(v);
    }
    __syncthreads();

    const int lane = threadIdx.x & 31;
    const int wave = threadIdx.x >> 5;
    const int ct   = wave & 3;        // cout tile (16 rows)
    const int ph   = wave >> 2;       // pixel half (64 pixels)
    const int l15  = lane & 15;
    const int hi   = lane >> 4;       // wave half per WMMA layout

    v8f acc[4] = {v8f{}, v8f{}, v8f{}, v8f{}};

    union Frag { v16bf v; v4u q[2]; };

    // per-lane A row base: reordered weights W_b[cout_row][576]
    const unsigned short* wb = wr + (size_t)(b * COUT_ + ct * 16 + l15) * KTOT;

    // A fragment: lane<16 holds K {0..15}, lane>=16 {16..31}
    auto loadA = [&](Frag& f, int s) {
        const int ka = s * 32 + hi * 16;
        f.q[0] = *(const v4u*)(wb + ka);
        f.q[1] = *(const v4u*)(wb + ka + 8);
    };
    // B fragments: 4 pixel tiles, 16 contiguous ci per lane (32B LDS read)
    auto loadB = [&](Frag* f, int s) {
        const int tap = s >> 1;                 // same (kh,kw) for 2 steps
        const int kh  = tap / 3;
        const int kw  = tap - kh * 3;
        const int cil = ((s & 1) << 5) + hi * 16;
        const unsigned short* brow = &xs[(kh * 130 + kw) * 64 + cil];
#pragma unroll
        for (int j = 0; j < 4; ++j) {
            const int p = ph * 64 + j * 16 + l15;   // output pixel (column N)
            f[j].v = *(const v16bf*)(brow + p * 64);
        }
    };

    Frag a_cur, a_nxt;
    Frag b_cur[4], b_nxt[4];
    loadA(a_cur, 0);
    loadB(b_cur, 0);

#pragma unroll
    for (int s = 0; s < 18; ++s) {             // K-steps of 32
        if (s + 1 < 18) {                      // prefetch next step
            loadA(a_nxt, s + 1);
            loadB(b_nxt, s + 1);
        }
#pragma unroll
        for (int j = 0; j < 4; ++j) {
            acc[j] = __builtin_amdgcn_wmma_f32_16x16x32_bf16(
                false, a_cur.v, false, b_cur[j].v, (short)0, acc[j], false, false);
        }
        if (s + 1 < 18) {                      // rotate (renamed away by unroll)
            a_cur = a_nxt;
#pragma unroll
            for (int j = 0; j < 4; ++j) b_cur[j] = b_nxt[j];
        }
    }

    // ---- store: lane<16 -> M=r, lane>=16 -> M=r+8 ; N = l15 ----
    float bias[8];
#pragma unroll
    for (int r = 0; r < 8; ++r) bias[r] = conv_b[ct * 16 + hi * 8 + r];

#pragma unroll
    for (int j = 0; j < 4; ++j) {
        const int p = ph * 64 + j * 16 + l15;
#pragma unroll
        for (int r = 0; r < 8; ++r) {
            const int m = ct * 16 + hi * 8 + r;
            out[(((size_t)b * COUT_ + m) * H_ + y) * W_ + p] = acc[j][r] + bias[r];
        }
    }
}

// ---------------------------------------------------------------------------
extern "C" void kernel_launch(void* const* d_in, const int* in_sizes, int n_in,
                              void* d_out, int out_size, void* d_ws, size_t ws_size,
                              hipStream_t stream) {
    const float* x      = (const float*)d_in[0];
    const float* c      = (const float*)d_in[1];
    const float* conv_w = (const float*)d_in[2];
    const float* conv_b = (const float*)d_in[3];
    const float* w1     = (const float*)d_in[4];
    const float* b1     = (const float*)d_in[5];
    const float* w2     = (const float*)d_in[6];
    const float* b2     = (const float*)d_in[7];

    unsigned short* wr = (unsigned short*)d_ws;   // 32*64*576 bf16 = 2.25 MB

    caconv_wgen<<<dim3(B_ * COUT_), dim3(192), 0, stream>>>(
        c, conv_w, w1, b1, w2, b2, wr);

    caconv_main<<<dim3(H_, B_), dim3(256), 0, stream>>>(
        x, wr, conv_b, (float*)d_out);
}
